// HConv2D_67010079752688
// MI455X (gfx1250) — matrix operand order repaired and checked by reading, not compiled, not run on hardware
//
#include <hip/hip_runtime.h>
#include <stdint.h>

typedef float    v8f   __attribute__((ext_vector_type(8)));
typedef __bf16   v16bf __attribute__((ext_vector_type(16)));
typedef uint32_t u32x4 __attribute__((ext_vector_type(4)));
typedef uint32_t u32x8 __attribute__((ext_vector_type(8)));

#define NB  16
#define HH  128
#define WW  128
#define CIN 128
#define FO  256

#define ROWSTRIDE (130 * CIN)   // one staged input row: cols -1..128, ci-major

__device__ __forceinline__ uint16_t f32_to_bf16(float f) {
  union { float f; uint32_t u; } x; x.f = f;
  uint32_t u = x.u;
  uint32_t r = u + 0x7FFFu + ((u >> 16) & 1u);      // round-to-nearest-even
  if ((u & 0x7F800000u) == 0x7F800000u) r = u;      // inf/nan passthrough
  return (uint16_t)(r >> 16);
}

// --- Pass 1: x fp32 -> bf16 (enables byte-exact TDM staging into LDS) ---
__global__ void prep_x(const float* __restrict__ x, uint16_t* __restrict__ x16, int n) {
  int i = blockIdx.x * blockDim.x + threadIdx.x;
  const int stride = gridDim.x * blockDim.x;
  for (; i < n; i += stride) x16[i] = f32_to_bf16(x[i]);
}

// --- Pass 2: build parity kernels (hex mask + roll) pre-swizzled into WMMA
// B-fragment order: [p][r][c][ftile(16f)][ktile(32ci)][lane(32)][j(16)]
// B layout (contiguous k order used here): lane = 16*g + fl ; N = fl ; K = 16*g + j.
__global__ void prep_w(const float* __restrict__ k, uint16_t* __restrict__ wb) {
  int idx = blockIdx.x * 256 + threadIdx.x;   // exactly 589824 threads
  int t = idx;
  const int j    = t & 15;  t >>= 4;
  const int lane = t & 31;  t >>= 5;
  const int kt   = t & 3;   t >>= 2;
  const int ft   = t & 15;  t >>= 4;
  const int c    = t % 3;   t /= 3;
  const int r    = t % 3;   t /= 3;
  const int p    = t;
  const int g  = lane >> 4;
  const int fl = lane & 15;
  const int f  = ft * 16 + fl;
  const int ci = kt * 32 + g * 16 + j;

  float val = 0.f;
  if (p == 0) {
    // even rows: kernel with mask[r!=center][col 0] = 0   (center=1, shift=-1)
    val = k[(((size_t)r * 3 + c) * CIN + ci) * FO + f];
    if (r != 1 && c == 0) val = 0.f;
  } else {
    // odd rows: Wp[1][0]=roll(k[0]), Wp[1][1]=roll(k[2]), Wp[1][2]=0 ; roll -1 on width
    if (r < 2 && c < 2) {
      const int sr = (r == 0) ? 0 : 2;
      val = k[(((size_t)sr * 3 + (c + 1)) * CIN + ci) * FO + f];
    }
  }
  wb[idx] = f32_to_bf16(val);
}

// --- Pass 3: implicit-GEMM hex conv. One block per (h, n, 64-filter tile).
// 8 waves: 4 (M=32 w-positions each) x 2 (N=32 filters each).
// Double-buffered TDM row staging: prefetch row r+1 while computing row r.
__global__ __launch_bounds__(256) void hconv_wmma(
    const uint16_t* __restrict__ x16, const uint16_t* __restrict__ wb,
    const float* __restrict__ bias, float* __restrict__ out) {
  extern __shared__ uint16_t xs[];       // 2 * ROWSTRIDE bf16 = 66,560 B dynamic LDS

  const int h   = blockIdx.x;
  const int n   = blockIdx.y;
  const int f64 = blockIdx.z;            // 0..3 -> filters f64*64..+63
  const int par = h & 1;

  const int tid  = threadIdx.x;
  const int lane = tid & 31;
  const int wave = tid >> 5;
  const int wm   = wave & 3;             // M block of 32
  const int wn   = wave >> 2;            // N block of 32
  const int g    = lane >> 4;
  const int fl   = lane & 15;

  v8f acc00 = {}, acc01 = {}, acc10 = {}, acc11 = {};

  // zero halo columns (x col -1 and x col 128) of both buffers
  if (tid < CIN) {
    xs[tid] = 0;             xs[129 * CIN + tid] = 0;
    xs[ROWSTRIDE + tid] = 0; xs[ROWSTRIDE + 129 * CIN + tid] = 0;
  }

  const uint32_t ldsb = (uint32_t)(uintptr_t)(&xs[0]);  // low 32b of flat = LDS byte addr

  // TDM issue: 2D tile [ci=128][cols=128] -> LDS cols 1..128 of buffer `buf`;
  // rows past H are zero-filled by forcing tensor_dim1 = 0 (OOB reads -> 0).
  auto tdm_issue = [&](int row, int buf) {
    const int rc  = row < HH ? row : (HH - 1);
    const uint64_t ga = (uint64_t)(uintptr_t)(x16 + ((size_t)n * HH + rc) * WW * CIN);
    const uint32_t dim1 = (row < HH) ? (uint32_t)WW : 0u;
    u32x4 d0;
    d0[0] = 1u;                                                 // count=1, user descriptor
    d0[1] = ldsb + (uint32_t)((buf * ROWSTRIDE + CIN) * 2);     // lds_addr: skip halo col 0
    d0[2] = (uint32_t)ga;                                       // global_addr[31:0]
    d0[3] = ((uint32_t)(ga >> 32) & 0x01FFFFFFu) | 0x80000000u; // addr[56:32] | type=2
    u32x8 d1;
    d1[0] = 0x00010000u;                 // data_size = 2 bytes
    d1[1] = ((uint32_t)CIN) << 16;       // tensor_dim0 = 128 (ci)
    d1[2] = dim1 << 16;                  // tensor_dim1 (cols); 0 => zero-fill
    d1[3] = ((uint32_t)CIN) << 16;       // tile_dim0 = 128
    d1[4] = (uint32_t)WW;                // tile_dim1 = 128
    d1[5] = (uint32_t)CIN;               // tensor_dim0_stride = 128 elems / col
    d1[6] = 0u;
    d1[7] = 0u;
    asm volatile("tensor_load_to_lds %0, %1" :: "s"(d0), "s"(d1) : "memory");
  };

  __syncthreads();                        // halo zeros visible before TDM data is used
  if (wave == 0) tdm_issue(h, 0);         // L0 -> buf0

#pragma unroll
  for (int r = 0; r < 3; ++r) {
    if (r > 0) __syncthreads();           // compute r-1 done -> buf[(r+1)&1] is free
    if (wave == 0) {
      if (r < 2) {
        tdm_issue(h + r + 1, (r + 1) & 1);          // prefetch next row
        __builtin_amdgcn_s_wait_tensorcnt(1);       // row r done; row r+1 in flight
      } else {
        __builtin_amdgcn_s_wait_tensorcnt(0);       // last row done
      }
    }
    __syncthreads();                      // staged row r visible to all waves

    const uint16_t* xb  = xs + (r & 1) * ROWSTRIDE;
    const uint16_t* wbp = wb + (((size_t)par * 3 + r) * 3) * (16 * 4 * 512);
#pragma unroll
    for (int c = 0; c < 3; ++c) {
#pragma unroll
      for (int kt = 0; kt < 4; ++kt) {
        // A fragments (16x32 bf16), contiguous k order: lane holds M=fl,
        // K = {16g..16g+15}  -> two ds_load_b128 each
        union { u32x4 q[2]; v16bf v; } A0, A1, B0, B1;
        {
          const int cib = kt * 32 + g * 16;
          const int m0  = wm * 32 + fl;
          const uint16_t* a0 = &xb[(m0 + c) * CIN + cib];
          A0.q[0] = *(const u32x4*)(a0);
          A0.q[1] = *(const u32x4*)(a0 + 8);
          const uint16_t* a1 = &xb[(m0 + 16 + c) * CIN + cib];
          A1.q[0] = *(const u32x4*)(a1);
          A1.q[1] = *(const u32x4*)(a1 + 8);
        }
        // B fragments: pre-swizzled contiguous 1KB frags, L2-resident
        {
          const int ft0 = f64 * 4 + wn * 2;
          const uint16_t* b0 = wbp + (((size_t)c * 16 + ft0) * 4 + kt) * 512 + lane * 16;
          B0.q[0] = *(const u32x4*)(b0);
          B0.q[1] = *(const u32x4*)(b0 + 8);
          const uint16_t* b1 = wbp + (((size_t)c * 16 + ft0 + 1) * 4 + kt) * 512 + lane * 16;
          B1.q[0] = *(const u32x4*)(b1);
          B1.q[1] = *(const u32x4*)(b1 + 8);
        }
        acc00 = __builtin_amdgcn_wmma_f32_16x16x32_bf16(false, A0.v, false, B0.v, (short)0, acc00, false, false);
        acc01 = __builtin_amdgcn_wmma_f32_16x16x32_bf16(false, A0.v, false, B1.v, (short)0, acc01, false, false);
        acc10 = __builtin_amdgcn_wmma_f32_16x16x32_bf16(false, A1.v, false, B0.v, (short)0, acc10, false, false);
        acc11 = __builtin_amdgcn_wmma_f32_16x16x32_bf16(false, A1.v, false, B1.v, (short)0, acc11, false, false);
      }
    }
  }

  // Epilogue: D layout — VGPR q holds M = q + 8g, N = fl. Bias + ReLU, fp32 out.
  const int fbase = f64 * 64 + wn * 32;
  const float b0 = bias[fbase + fl];
  const float b1 = bias[fbase + 16 + fl];
  float* outp = out + ((size_t)n * HH + h) * WW * FO;
#pragma unroll
  for (int q = 0; q < 8; ++q) {
    const int m0 = wm * 32 + 8 * g + q;
    const int m1 = m0 + 16;
    float v;
    v = acc00[q] + b0; outp[(size_t)m0 * FO + fbase + fl]      = v > 0.f ? v : 0.f;
    v = acc01[q] + b1; outp[(size_t)m0 * FO + fbase + 16 + fl] = v > 0.f ? v : 0.f;
    v = acc10[q] + b0; outp[(size_t)m1 * FO + fbase + fl]      = v > 0.f ? v : 0.f;
    v = acc11[q] + b1; outp[(size_t)m1 * FO + fbase + 16 + fl] = v > 0.f ? v : 0.f;
  }
}

extern "C" void kernel_launch(void* const* d_in, const int* in_sizes, int n_in,
                              void* d_out, int out_size, void* d_ws, size_t ws_size,
                              hipStream_t stream) {
  const float* x    = (const float*)d_in[0];   // (16,128,128,128) f32
  const float* kern = (const float*)d_in[1];   // (3,3,128,256) f32
  const float* bias = (const float*)d_in[2];   // (256,) f32
  float* out = (float*)d_out;                  // (16,128,128,256) f32

  // workspace: [0, 64MB) x in bf16 ; then 1.125MB swizzled weights
  uint16_t* x16 = (uint16_t*)d_ws;
  uint16_t* wb  = (uint16_t*)((char*)d_ws + (size_t)NB * HH * WW * CIN * sizeof(uint16_t));

  prep_x<<<4096, 256, 0, stream>>>(x, x16, NB * HH * WW * CIN);
  prep_w<<<2304, 256, 0, stream>>>(kern, wb);                 // 589824 elements
  hconv_wmma<<<dim3(HH, NB, FO / 64), 256,
               2 * ROWSTRIDE * sizeof(uint16_t), stream>>>(x16, wb, bias, out);
}
